// GNNModel_87333864997649
// MI455X (gfx1250) — matrix operand order repaired and checked by reading, not compiled, not run on hardware
//
#include <hip/hip_runtime.h>
#include <hip/hip_bf16.h>
#include <cstddef>

// Problem constants (from reference)
#define BDIM   512
#define LSEQ   100
#define HDIM   256
#define VDIM   100000
#define NPG    40
#define EPG    100
#define NNODE  (BDIM * NPG)     // 20480
#define NEDGE  (BDIM * EPG)     // 51200
#define H2DIM  128
#define NENT   (NEDGE + NNODE)  // 71680 (edges + self loops)

typedef __attribute__((ext_vector_type(16))) __bf16 v16bf;
typedef __attribute__((ext_vector_type(8)))  float  v8f;

union Frag { unsigned u[8]; v16bf v; };

// ---------------- bf16 packing helpers ----------------
__device__ __forceinline__ unsigned bfr(float f) {
  union { float f; unsigned u; } c; c.f = f;
  return c.u + 0x8000u;                      // round-to-nearest (ties away)
}
// pack two floats into one dword of bf16 pair: low16 = lo, high16 = hi
__device__ __forceinline__ unsigned pack2(float lo, float hi) {
  return __builtin_amdgcn_perm(bfr(hi), bfr(lo), 0x07060302u);
}
__device__ __forceinline__ float4 ld4(const float* p) { return *(const float4*)p; }
__device__ __forceinline__ float4 sub4(float4 a, float4 b) { a.x-=b.x; a.y-=b.y; a.z-=b.z; a.w-=b.w; return a; }

__device__ __forceinline__ v16bf pack_frag(float4 f0, float4 f1, float4 f2, float4 f3) {
  Frag r;
  r.u[0]=pack2(f0.x,f0.y); r.u[1]=pack2(f0.z,f0.w);
  r.u[2]=pack2(f1.x,f1.y); r.u[3]=pack2(f1.z,f1.w);
  r.u[4]=pack2(f2.x,f2.y); r.u[5]=pack2(f2.z,f2.w);
  r.u[6]=pack2(f3.x,f3.y); r.u[7]=pack2(f3.z,f3.w);
  return r.v;
}
// A fragment (16x32 MxK, ISA layout) from fp32: p = rowBase + k0 + hi*8
__device__ __forceinline__ v16bf a_frag1(const float* p) {
  return pack_frag(ld4(p), ld4(p + 4), ld4(p + 16), ld4(p + 20));
}
__device__ __forceinline__ v16bf a_frag_diff(const float* pa, const float* pb) {
  return pack_frag(sub4(ld4(pa),ld4(pb)),       sub4(ld4(pa+4),ld4(pb+4)),
                   sub4(ld4(pa+16),ld4(pb+16)), sub4(ld4(pa+20),ld4(pb+20)));
}
// A fragment from pre-packed bf16 dwords: p = rowDw + (k0>>1) + hi*4
__device__ __forceinline__ v16bf a_frag_packed(const unsigned* p) {
  Frag r;
  const uint4 q0 = *(const uint4*)p;
  const uint4 q1 = *(const uint4*)(p + 8);
  r.u[0]=q0.x; r.u[1]=q0.y; r.u[2]=q0.z; r.u[3]=q0.w;
  r.u[4]=q1.x; r.u[5]=q1.y; r.u[6]=q1.z; r.u[7]=q1.w;
  return r.v;
}
// B fragment from pre-packed bf16 dwords: p = colRowDw + (k0>>1) + hi*4
__device__ __forceinline__ v16bf b_frag_packed(const unsigned* p) {
  Frag r;
  const uint4 q0 = *(const uint4*)p;
  const uint4 q1 = *(const uint4*)(p + 8);
  r.u[0]=q0.x; r.u[1]=q0.y; r.u[2]=q0.z; r.u[3]=q0.w;
  r.u[4]=q1.x; r.u[5]=q1.y; r.u[6]=q1.z; r.u[7]=q1.w;
  return r.v;
}

__device__ __forceinline__ float sigmoidf_(float x) { return 1.0f / (1.0f + __expf(-x)); }

// monotonic float<->uint mapping for atomicMax on floats
__device__ __forceinline__ unsigned ordf(float f) {
  union { float f; unsigned u; } c; c.f = f;
  return (c.u & 0x80000000u) ? ~c.u : (c.u | 0x80000000u);
}
__device__ __forceinline__ float unordf(unsigned u) {
  union { float f; unsigned u; } c;
  c.u = (u & 0x80000000u) ? (u ^ 0x80000000u) : ~u;
  return c.f;
}

// ---------------- fill / pack kernels ----------------
__global__ void k_fill_f32(float* p, float v, size_t n) {
  size_t i = (size_t)blockIdx.x * blockDim.x + threadIdx.x;
  if (i < n) p[i] = v;
}
__global__ void k_fill_ordneg(unsigned* p, size_t n) {
  size_t i = (size_t)blockIdx.x * blockDim.x + threadIdx.x;
  if (i < n) p[i] = ordf(-1e30f);
}
// Pack LSTM weights (all 4 layers): Wbf[l][col][kp] = bf16{Wc[col][2kp], Wc[col][2kp+1]},
// Wc = [wih | whh] along K (K = 512)
__global__ void k_pack_lstm_w(const float* __restrict__ wih, const float* __restrict__ whh,
                              unsigned* __restrict__ Wbf) {
  const int idx = blockIdx.x * blockDim.x + threadIdx.x;   // 4*1024*256
  if (idx >= 4 * 1024 * 256) return;
  const int kp = idx & 255;
  const int col = (idx >> 8) & 1023;
  const int l = idx >> 18;
  const int k = kp << 1;
  const float* w = (k < HDIM)
      ? (wih + (size_t)l * 1024 * HDIM + (size_t)col * HDIM + k)
      : (whh + (size_t)l * 1024 * HDIM + (size_t)col * HDIM + (k - HDIM));
  Wbf[idx] = pack2(w[0], w[1]);
}
// Transpose-pack fp32 Bm[K][N] -> bf16 Bt[n][K/2] dwords
__global__ void k_pack_bt(const float* __restrict__ Bm, unsigned* __restrict__ Bt,
                          int K, int N) {
  const int Kd = K >> 1;
  const int idx = blockIdx.x * blockDim.x + threadIdx.x;
  if (idx >= N * Kd) return;
  const int kp = idx % Kd, n = idx / Kd;
  const int k = kp << 1;
  Bt[idx] = pack2(Bm[(size_t)k * N + n], Bm[(size_t)(k + 1) * N + n]);
}
// Pack scaled vocab A: Abf[row][kp] over K=512: uw*dyn || (1-uw)*pooled
__global__ void k_pack_a(const float* __restrict__ dyn, const float* __restrict__ pooled,
                         const float* __restrict__ uwv, unsigned* __restrict__ Abf) {
  const int idx = blockIdx.x * blockDim.x + threadIdx.x;   // 512*256
  if (idx >= BDIM * 256) return;
  const int kp = idx & 255, row = idx >> 8;
  const int k = kp << 1;
  float s, v0, v1;
  if (k < HDIM) {
    s = uwv[row];
    v0 = dyn[(size_t)row * HDIM + k]; v1 = dyn[(size_t)row * HDIM + k + 1];
  } else {
    s = 1.0f - uwv[row];
    v0 = pooled[(size_t)row * HDIM + k - HDIM]; v1 = pooled[(size_t)row * HDIM + k - HDIM + 1];
  }
  Abf[idx] = pack2(s * v0, s * v1);
}

// ---------------- LSTM ----------------
// G[b, 0:4H] = [x_t | h] @ Wc^T + (bih + bhh); Wc pre-packed bf16, h pre-packed bf16
// MODE 0: x_t = emb[idx[row*stride + t]];  MODE 1: x_t = xA - xB
template <int MODE>
__global__ void __launch_bounds__(256)
k_lstm_gemm(const float* __restrict__ emb, const int* __restrict__ idx, int seqStride, int t,
            const float* __restrict__ xA, const float* __restrict__ xB,
            const unsigned* __restrict__ hbf,
            const unsigned* __restrict__ Wbf,
            const float* __restrict__ bih, const float* __restrict__ bhh,
            float* __restrict__ G) {
  const int tile = blockIdx.x * 8 + (threadIdx.x >> 5);  // 2048 tiles: 32 M x 64 N
  const int m0 = (tile & 31) << 4;
  const int n0 = (tile >> 5) << 4;
  const int lane = threadIdx.x & 31;
  const int hi = lane >> 4;
  const int row = m0 + (lane & 15);
  const int col = n0 + (lane & 15);
  const float* xr0;
  const float* xr1 = nullptr;
  if (MODE == 0) {
    xr0 = emb + (size_t)idx[row * seqStride + t] * HDIM;
  } else {
    xr0 = xA + (size_t)row * HDIM;
    xr1 = xB + (size_t)row * HDIM;
  }
  const unsigned* hrow = hbf + (size_t)row * 128;
  const unsigned* wrow = Wbf + (size_t)col * 256;
  v8f acc = {};
#pragma unroll
  for (int kb = 0; kb < 8; ++kb) {         // x half (K 0..255), fp32 -> bf16 on the fly
    const int k0 = kb << 5;
    v16bf a = (MODE == 0) ? a_frag1(xr0 + k0 + hi * 8)
                          : a_frag_diff(xr0 + k0 + hi * 8, xr1 + k0 + hi * 8);
    v16bf b = b_frag_packed(wrow + (k0 >> 1) + hi * 4);
    acc = __builtin_amdgcn_wmma_f32_16x16x32_bf16(false, a, false, b, (short)0, acc, false, false);
  }
#pragma unroll
  for (int kb = 0; kb < 8; ++kb) {         // h half (K 256..511), fully pre-packed
    const int k0 = kb << 5;
    v16bf a = a_frag_packed(hrow + (k0 >> 1) + hi * 4);
    v16bf b = b_frag_packed(wrow + 128 + (k0 >> 1) + hi * 4);
    acc = __builtin_amdgcn_wmma_f32_16x16x32_bf16(false, a, false, b, (short)0, acc, false, false);
  }
  const float bias = bih[col] + bhh[col];
  const int mbase = m0 + (hi << 3);
#pragma unroll
  for (int v = 0; v < 8; ++v)
    G[(size_t)(mbase + v) * (4 * HDIM) + col] = acc[v] + bias;
}

// gates + state update; also emits packed bf16 h for the next step's GEMM
__global__ void k_lstm_pw(const float* __restrict__ G, const float* __restrict__ cin,
                          float* __restrict__ cout, float* __restrict__ hout,
                          unsigned* __restrict__ hbf, float* __restrict__ ys) {
  const int idx = blockIdx.x * blockDim.x + threadIdx.x;   // BDIM*HDIM/2
  if (idx >= BDIM * HDIM / 2) return;
  const int b = idx >> 7;
  const int hh = (idx & 127) << 1;
  const float* g = G + (size_t)b * 4 * HDIM;
  float h2[2], c2[2];
#pragma unroll
  for (int j = 0; j < 2; ++j) {
    const int q = hh + j;
    const float gi = sigmoidf_(g[q]);
    const float gf = sigmoidf_(g[HDIM + q]);
    const float gg = tanhf(g[2 * HDIM + q]);
    const float go = sigmoidf_(g[3 * HDIM + q]);
    const float c = gf * cin[(size_t)b * HDIM + q] + gi * gg;
    c2[j] = c;
    h2[j] = go * tanhf(c);
  }
  const size_t base = (size_t)b * HDIM + hh;
  cout[base] = c2[0]; cout[base + 1] = c2[1];
  hout[base] = h2[0]; hout[base + 1] = h2[1];
  hbf[(size_t)b * 128 + (hh >> 1)] = pack2(h2[0], h2[1]);
  if (ys) { ys[base] = h2[0]; ys[base + 1] = h2[1]; }
}

__global__ void k_layernorm(const float* h0, const float* h1, const float* h2, const float* h3,
                            const float* sc, const float* bi, float* dyn) {
  const int b = blockIdx.x, t = threadIdx.x;
  __shared__ float red[HDIM];
  const size_t i = (size_t)b * HDIM + t;
  const float v = h0[i] + h1[i] + h2[i] + h3[i];
  red[t] = v; __syncthreads();
  for (int s = HDIM / 2; s > 0; s >>= 1) { if (t < s) red[t] += red[t + s]; __syncthreads(); }
  const float mu = red[0] * (1.0f / HDIM); __syncthreads();
  const float d = v - mu;
  red[t] = d * d; __syncthreads();
  for (int s = HDIM / 2; s > 0; s >>= 1) { if (t < s) red[t] += red[t + s]; __syncthreads(); }
  const float var = red[0] * (1.0f / HDIM);
  dyn[i] = sc[t] * d * rsqrtf(var + 1e-5f) + bi[t];
}

__global__ void k_mse(const float* dyn, const float* hreal, float* out) {
  __shared__ float red[256];
  float acc = 0.f;
  for (int i = threadIdx.x; i < BDIM * HDIM; i += 256) {
    const float d = dyn[i] - hreal[i]; acc += d * d;
  }
  red[threadIdx.x] = acc; __syncthreads();
  for (int s = 128; s > 0; s >>= 1) { if (threadIdx.x < s) red[threadIdx.x] += red[threadIdx.x + s]; __syncthreads(); }
  if (threadIdx.x == 0) out[0] = red[0] * (1.0f / (BDIM * HDIM));
}

// ---------------- GAT ----------------
__global__ void k_gather(const int* __restrict__ x, const float* __restrict__ gemb,
                         float* __restrict__ nodes) {
  const int idx = blockIdx.x * blockDim.x + threadIdx.x;
  if (idx >= NNODE * HDIM) return;
  const int node = idx >> 8, k = idx & 255;
  nodes[idx] = gemb[(size_t)(x[node] - 1) * HDIM + k];
}

// C = act([A0|A1] @ B + bias); B pre-packed as Bt[n][K/2] bf16 dwords
template <int KH>   // number of 256-wide K halves (1 or 2; A1 used when KH==2)
__global__ void __launch_bounds__(256)
k_gemm_kn(const float* __restrict__ A0, const float* __restrict__ A1,
          const unsigned* __restrict__ Bt, const float* __restrict__ bias,
          float* __restrict__ C, int M, int Ncols, int act) {
  const int ntTiles = Ncols >> 4;
  const int tile = blockIdx.x * 8 + (threadIdx.x >> 5);
  if (tile >= (M >> 4) * ntTiles) return;   // wave-uniform
  const int m0 = (tile / ntTiles) << 4;
  const int n0 = (tile % ntTiles) << 4;
  const int lane = threadIdx.x & 31;
  const int hi = lane >> 4;
  const int row = m0 + (lane & 15);
  const int col = n0 + (lane & 15);
  const float* ar0 = A0 + (size_t)row * HDIM;
  const unsigned* brow = Bt + (size_t)col * (KH * 128);
  v8f acc = {};
#pragma unroll
  for (int kb = 0; kb < 8; ++kb) {
    const int k0 = kb << 5;
    v16bf a = a_frag1(ar0 + k0 + hi * 8);
    v16bf b = b_frag_packed(brow + (k0 >> 1) + hi * 4);
    acc = __builtin_amdgcn_wmma_f32_16x16x32_bf16(false, a, false, b, (short)0, acc, false, false);
  }
  if (KH == 2) {
    const float* ar1 = A1 + (size_t)row * HDIM;
#pragma unroll
    for (int kb = 0; kb < 8; ++kb) {
      const int k0 = kb << 5;
      v16bf a = a_frag1(ar1 + k0 + hi * 8);
      v16bf b = b_frag_packed(brow + 128 + (k0 >> 1) + hi * 4);
      acc = __builtin_amdgcn_wmma_f32_16x16x32_bf16(false, a, false, b, (short)0, acc, false, false);
    }
  }
  const float bv = bias ? bias[col] : 0.f;
  const int mbase = m0 + (hi << 3);
#pragma unroll
  for (int v = 0; v < 8; ++v) {
    float r = acc[v] + bv;
    if (act == 1) r = tanhf(r);
    else if (act == 2) r = sigmoidf_(r);
    C[(size_t)(mbase + v) * Ncols + col] = r;
  }
}

__global__ void k_gat_ae(const float* __restrict__ xw, const float* __restrict__ asrc,
                         const float* __restrict__ adst, float* __restrict__ aes,
                         float* __restrict__ aed) {
  const int idx = blockIdx.x * blockDim.x + threadIdx.x;
  if (idx >= NNODE * 2) return;
  const int i = idx >> 1, hd = idx & 1;
  const float* xr = xw + (size_t)i * HDIM + hd * H2DIM;
  const float* as = asrc + hd * H2DIM;
  const float* ad = adst + hd * H2DIM;
  float s = 0.f, d = 0.f;
  for (int j = 0; j < H2DIM; ++j) { s += xr[j] * as[j]; d += xr[j] * ad[j]; }
  aes[idx] = s; aed[idx] = d;
}

__device__ __forceinline__ float leaky02(float v) { return v > 0.f ? v : 0.2f * v; }

__global__ void k_edge_max(const int* __restrict__ src, const int* __restrict__ dst,
                           const float* __restrict__ aes, const float* __restrict__ aed,
                           unsigned* __restrict__ mh) {
  const int e = blockIdx.x * blockDim.x + threadIdx.x;
  if (e >= NENT) return;
  const int si = (e < NEDGE) ? src[e] : (e - NEDGE);
  const int di = (e < NEDGE) ? dst[e] : (e - NEDGE);
#pragma unroll
  for (int hd = 0; hd < 2; ++hd) {
    const float v = leaky02(aes[si * 2 + hd] + aed[di * 2 + hd]);
    atomicMax(&mh[di * 2 + hd], ordf(v));
  }
}

__global__ void k_edge_ex(const int* __restrict__ src, const int* __restrict__ dst,
                          const float* __restrict__ aes, const float* __restrict__ aed,
                          const unsigned* __restrict__ mh, float* __restrict__ exb,
                          float* __restrict__ sh) {
  const int e = blockIdx.x * blockDim.x + threadIdx.x;
  if (e >= NENT) return;
  const int si = (e < NEDGE) ? src[e] : (e - NEDGE);
  const int di = (e < NEDGE) ? dst[e] : (e - NEDGE);
#pragma unroll
  for (int hd = 0; hd < 2; ++hd) {
    const float v = leaky02(aes[si * 2 + hd] + aed[di * 2 + hd]);
    const float ex = __expf(v - unordf(mh[di * 2 + hd]));
    exb[e * 2 + hd] = ex;
    atomicAdd(&sh[di * 2 + hd], ex);
  }
}

__global__ void k_conv_acc(const int* __restrict__ src, const int* __restrict__ dst,
                           const float* __restrict__ exb, const float* __restrict__ sh,
                           const float* __restrict__ xw, float* __restrict__ conv) {
  const int lane = threadIdx.x & 31;
  const int e = blockIdx.x * 8 + (threadIdx.x >> 5);
  if (e >= NENT) return;
  const int si = (e < NEDGE) ? src[e] : (e - NEDGE);
  const int di = (e < NEDGE) ? dst[e] : (e - NEDGE);
  const float a0 = exb[e * 2]     / sh[di * 2];
  const float a1 = exb[e * 2 + 1] / sh[di * 2 + 1];
#pragma unroll
  for (int j = 0; j < 8; ++j) {
    const int h = lane + 32 * j;
    const float al = (h < H2DIM) ? a0 : a1;
    atomicAdd(&conv[(size_t)di * HDIM + h], al * xw[(size_t)si * HDIM + h]);
  }
}

__global__ void k_add_bias(float* __restrict__ p, const float* __restrict__ b, int n) {
  const int idx = blockIdx.x * blockDim.x + threadIdx.x;
  if (idx < n) p[idx] += b[idx & 255];
}

__global__ void k_combine(const float* __restrict__ gate, const float* __restrict__ inb,
                          float* __restrict__ outb, int n) {
  const int idx = blockIdx.x * blockDim.x + threadIdx.x;
  if (idx < n) {
    const float g = gate[idx];
    outb[idx] = g * outb[idx] + (1.0f - g) * inb[idx];
  }
}

// ---------------- pooling / fusion ----------------
__global__ void k_pool_g(const float* __restrict__ g1, const float* __restrict__ gw,
                         const float* __restrict__ gb, const int* __restrict__ hist,
                         float* __restrict__ gsc) {
  const int i = blockIdx.x * blockDim.x + threadIdx.x;
  if (i >= NNODE) return;
  float s = gb[0];
  for (int j = 0; j < H2DIM; ++j) s += g1[(size_t)i * H2DIM + j] * gw[j];
  gsc[i] = (hist[i] == 0) ? -1e9f : s;
}

__global__ void k_pooled(const float* __restrict__ gsc, const float* __restrict__ nodes,
                         float* __restrict__ pooled) {
  const int b = blockIdx.x;
  __shared__ float alpha[NPG];
  __shared__ float ssum;
  if (threadIdx.x == 0) {
    float mx = -1e30f;
    for (int j = 0; j < NPG; ++j) mx = fmaxf(mx, gsc[b * NPG + j]);
    float s = 0.f;
    for (int j = 0; j < NPG; ++j) { const float e = __expf(gsc[b * NPG + j] - mx); alpha[j] = e; s += e; }
    ssum = s;
  }
  __syncthreads();
  const float inv = 1.0f / ssum;
  const int h = threadIdx.x;
  float acc = 0.f;
  for (int j = 0; j < NPG; ++j) acc += alpha[j] * nodes[(size_t)(b * NPG + j) * HDIM + h];
  pooled[(size_t)b * HDIM + h] = acc * inv;
}

__global__ void k_uw(const float* __restrict__ dyn, const float* __restrict__ pooled,
                     const float* __restrict__ wlW, const float* __restrict__ wlB,
                     float* __restrict__ uwv) {
  const int b = blockIdx.x * blockDim.x + threadIdx.x;
  if (b >= BDIM) return;
  float s = wlB[0];
  for (int k = 0; k < HDIM; ++k) s += dyn[(size_t)b * HDIM + k] * wlW[k];
  for (int k = 0; k < HDIM; ++k) s += pooled[(size_t)b * HDIM + k] * wlW[HDIM + k];
  uwv[b] = sigmoidf_(s);
}

// final[b, v] = uw[b]*(dyn[b] . emb[v]) + (1-uw[b])*(pooled[b] . gemb[v])
// One block per 16-column vocab tile: stage 16x512 fp32 panel -> bf16 LDS once,
// then 8 waves x 4 M-tiles consume it with pre-packed A (zero conversion in loop).
__global__ void __launch_bounds__(256)
k_vocab(const unsigned* __restrict__ Abf, const float* __restrict__ emb,
        const float* __restrict__ gemb, float* __restrict__ out) {
  __shared__ unsigned lds[16][264];          // 16 rows x 512 bf16 (+pad), dword-packed
  const int n0 = blockIdx.x << 4;            // 6250 blocks
  // ---- stage ----
#pragma unroll
  for (int it = 0; it < 8; ++it) {
    const int f = threadIdx.x + (it << 8);   // float4 index in [0, 2048)
    const int n = f >> 7;                    // 128 float4 per row (K = 512)
    const int k = (f & 127) << 2;
    const float* sp = (k < HDIM) ? (emb  + (size_t)(n0 + n) * HDIM + k)
                                 : (gemb + (size_t)(n0 + n) * HDIM + (k - HDIM));
    const float4 x = ld4(sp);
    uint2 pr; pr.x = pack2(x.x, x.y); pr.y = pack2(x.z, x.w);
    *(uint2*)&lds[n][k >> 1] = pr;
  }
  __syncthreads();
  // ---- compute ----
  const int wave = threadIdx.x >> 5;
  const int lane = threadIdx.x & 31;
  const int hi = lane >> 4;
  const int col = n0 + (lane & 15);
  const unsigned* brow = &lds[lane & 15][hi << 2];
  for (int mi = 0; mi < 4; ++mi) {
    const int m0 = (wave + (mi << 3)) << 4;
    const int row = m0 + (lane & 15);
    const unsigned* arow = Abf + (size_t)row * 256;
    v8f acc = {};
#pragma unroll
    for (int kb = 0; kb < 16; ++kb) {        // K = 512
      const int k0 = kb << 5;
      v16bf a = a_frag_packed(arow + (k0 >> 1) + hi * 4);
      Frag bf;
      const uint4 q0 = *(const uint4*)(brow + (k0 >> 1));
      const uint4 q1 = *(const uint4*)(brow + (k0 >> 1) + 8);
      bf.u[0]=q0.x; bf.u[1]=q0.y; bf.u[2]=q0.z; bf.u[3]=q0.w;
      bf.u[4]=q1.x; bf.u[5]=q1.y; bf.u[6]=q1.z; bf.u[7]=q1.w;
      acc = __builtin_amdgcn_wmma_f32_16x16x32_bf16(false, a, false, bf.v, (short)0, acc, false, false);
    }
    const int mbase = m0 + (hi << 3);
#pragma unroll
    for (int v = 0; v < 8; ++v)
      out[(size_t)(mbase + v) * VDIM + col] = acc[v];
  }
}

// ---------------- host ----------------
extern "C" void kernel_launch(void* const* d_in, const int* in_sizes, int n_in,
                              void* d_out, int out_size, void* d_ws, size_t ws_size,
                              hipStream_t stream) {
  (void)in_sizes; (void)n_in; (void)out_size; (void)ws_size;
  const int*   seq   = (const int*)  d_in[0];
  const int*   yit   = (const int*)  d_in[1];
  const int*   xids  = (const int*)  d_in[2];
  const int*   eidx  = (const int*)  d_in[3];
  const int*   hist  = (const int*)  d_in[5];
  const float* emb   = (const float*)d_in[6];
  const float* gemb  = (const float*)d_in[7];
  const float* wih   = (const float*)d_in[8];
  const float* whh   = (const float*)d_in[9];
  const float* bih   = (const float*)d_in[10];
  const float* bhh   = (const float*)d_in[11];
  const float* lnS   = (const float*)d_in[12];
  const float* lnB   = (const float*)d_in[13];
  const float* gatW  = (const float*)d_in[14];
  const float* gatAs = (const float*)d_in[15];
  const float* gatAd = (const float*)d_in[16];
  const float* gatB  = (const float*)d_in[17];
  const float* gateW = (const float*)d_in[18];
  const float* fuseW = (const float*)d_in[19];
  const float* ph1W  = (const float*)d_in[20];
  const float* ph1B  = (const float*)d_in[21];
  const float* pgW   = (const float*)d_in[22];
  const float* pgB   = (const float*)d_in[23];
  const float* wlW   = (const float*)d_in[24];
  const float* wlB   = (const float*)d_in[25];
  const int* srcI = eidx;
  const int* dstI = eidx + NEDGE;
  float* fout = (float*)d_out;

  float* W = (float*)d_ws;
  size_t off = 0;
  auto alloc = [&](size_t n) { size_t r = off; off += n; return r; };
  const size_t BH = (size_t)BDIM * HDIM;
  const size_t NH = (size_t)NNODE * HDIM;
  float* ys0     = W + alloc(100 * BH);
  float* ys1     = W + alloc(99 * BH);
  float* ys2     = W + alloc(98 * BH);
  float* hcBase  = W + alloc(8 * BH);           // h0,c0,h1,c1,h2,c2,h3,c3
  float* hreal   = W + alloc(BH);
  float* creal   = W + alloc(BH);
  float* G       = W + alloc((size_t)BDIM * 4 * HDIM);
  float* dyn     = W + alloc(BH);
  float* bufA    = W + alloc(NH);
  float* bufB    = W + alloc(NH);
  float* xw      = W + alloc(NH);
  float* conv    = W + alloc(NH);
  float* aes     = W + alloc((size_t)NNODE * 2);
  float* aed     = W + alloc((size_t)NNODE * 2);
  unsigned* mh   = (unsigned*)(W + alloc((size_t)NNODE * 2));
  float* sh      = W + alloc((size_t)NNODE * 2);
  float* exb     = W + alloc((size_t)NENT * 2);
  float* g1      = W + alloc((size_t)NNODE * H2DIM);
  float* gsc     = W + alloc((size_t)NNODE);
  float* pooled  = W + alloc(BH);
  float* uwv     = W + alloc((size_t)BDIM);
  // packed bf16 buffers (dword counts == float counts)
  unsigned* Wbf    = (unsigned*)(W + alloc((size_t)4 * 1024 * 256));
  unsigned* hbfAll = (unsigned*)(W + alloc((size_t)4 * BDIM * 128));
  unsigned* Abf    = (unsigned*)(W + alloc((size_t)BDIM * 256));
  unsigned* gatWt  = (unsigned*)(W + alloc((size_t)2 * 256 * 128));
  unsigned* gateWt = (unsigned*)(W + alloc((size_t)2 * 256 * 256));
  unsigned* fuseWt = (unsigned*)(W + alloc((size_t)2 * 256 * 128));
  unsigned* ph1Wt  = (unsigned*)(W + alloc((size_t)128 * 128));

  float* hb[4], *cb[4];
  unsigned* hbf[4];
  for (int l = 0; l < 4; ++l) {
    hb[l] = hcBase + (2 * l) * BH;
    cb[l] = hcBase + (2 * l + 1) * BH;
    hbf[l] = hbfAll + (size_t)l * BDIM * 128;
  }

  // one-time packs + zero init
  k_fill_f32<<<(int)((8 * BH + 255) / 256), 256, 0, stream>>>(hcBase, 0.f, 8 * BH);
  k_fill_f32<<<(int)((4 * BDIM * 128 + 255) / 256), 256, 0, stream>>>((float*)hbfAll, 0.f,
                                                                      (size_t)4 * BDIM * 128);
  k_pack_lstm_w<<<4096, 256, 0, stream>>>(wih, whh, Wbf);
  for (int l = 0; l < 2; ++l) {
    k_pack_bt<<<128, 256, 0, stream>>>(gatW  + (size_t)l * HDIM * HDIM,     gatWt  + (size_t)l * 256 * 128, 256, 256);
    k_pack_bt<<<256, 256, 0, stream>>>(gateW + (size_t)l * 2 * HDIM * HDIM, gateWt + (size_t)l * 256 * 256, 512, 256);
    k_pack_bt<<<128, 256, 0, stream>>>(fuseW + (size_t)l * HDIM * HDIM,     fuseWt + (size_t)l * 256 * 128, 256, 256);
  }
  k_pack_bt<<<64, 256, 0, stream>>>(ph1W, ph1Wt, 256, 128);

  const int PWB = (int)(BH / 2 / 256);   // pointwise blocks
  auto bslice = [&](int l) { return (size_t)l * 4 * HDIM; };
  auto wbs    = [&](int l) { return Wbf + (size_t)l * 1024 * 256; };

  // layer 0 over the sequence (x_t gathered from emb via seq)
  for (int t = 0; t < LSEQ; ++t) {
    k_lstm_gemm<0><<<256, 256, 0, stream>>>(emb, seq, LSEQ, t, nullptr, nullptr, hbf[0],
        wbs(0), bih + bslice(0), bhh + bslice(0), G);
    k_lstm_pw<<<PWB, 256, 0, stream>>>(G, cb[0], cb[0], hb[0], hbf[0], ys0 + (size_t)t * BH);
  }
  // layers 1..3 over successive differences
  for (int t = 0; t < LSEQ - 1; ++t) {
    k_lstm_gemm<1><<<256, 256, 0, stream>>>(emb, nullptr, 0, 0,
        ys0 + (size_t)(t + 1) * BH, ys0 + (size_t)t * BH, hbf[1],
        wbs(1), bih + bslice(1), bhh + bslice(1), G);
    k_lstm_pw<<<PWB, 256, 0, stream>>>(G, cb[1], cb[1], hb[1], hbf[1], ys1 + (size_t)t * BH);
  }
  for (int t = 0; t < LSEQ - 2; ++t) {
    k_lstm_gemm<1><<<256, 256, 0, stream>>>(emb, nullptr, 0, 0,
        ys1 + (size_t)(t + 1) * BH, ys1 + (size_t)t * BH, hbf[2],
        wbs(2), bih + bslice(2), bhh + bslice(2), G);
    k_lstm_pw<<<PWB, 256, 0, stream>>>(G, cb[2], cb[2], hb[2], hbf[2], ys2 + (size_t)t * BH);
  }
  for (int t = 0; t < LSEQ - 3; ++t) {
    k_lstm_gemm<1><<<256, 256, 0, stream>>>(emb, nullptr, 0, 0,
        ys2 + (size_t)(t + 1) * BH, ys2 + (size_t)t * BH, hbf[3],
        wbs(3), bih + bslice(3), bhh + bslice(3), G);
    k_lstm_pw<<<PWB, 256, 0, stream>>>(G, cb[3], cb[3], hb[3], hbf[3], nullptr);
  }

  // dynamic = LayerNorm(hT + h1 + h2 + h3)
  k_layernorm<<<BDIM, HDIM, 0, stream>>>(hb[0], hb[1], hb[2], hb[3], lnS, lnB, dyn);
  // real_seq: one more layer-0 step on emb[y_items] from (hT, cT)
  k_lstm_gemm<0><<<256, 256, 0, stream>>>(emb, yit, 1, 0, nullptr, nullptr, hbf[0],
      wbs(0), bih + bslice(0), bhh + bslice(0), G);
  k_lstm_pw<<<PWB, 256, 0, stream>>>(G, cb[0], creal, hreal, hbf[0], nullptr);
  k_mse<<<1, 256, 0, stream>>>(dyn, hreal, fout + (size_t)BDIM * VDIM);

  // ---- GAT stack ----
  k_gather<<<(int)(NH / 256), 256, 0, stream>>>(xids, gemb, bufA);
  for (int l = 0; l < 2; ++l) {
    const float* in = l ? bufB : bufA;
    float* out = l ? bufA : bufB;
    k_gemm_kn<1><<<2560, 256, 0, stream>>>(in, nullptr, gatWt + (size_t)l * 256 * 128,
                                           nullptr, xw, NNODE, HDIM, 0);
    k_gat_ae<<<160, 256, 0, stream>>>(xw, gatAs + (size_t)l * 2 * H2DIM,
                                      gatAd + (size_t)l * 2 * H2DIM, aes, aed);
    k_fill_ordneg<<<160, 256, 0, stream>>>(mh, (size_t)NNODE * 2);
    k_fill_f32<<<160, 256, 0, stream>>>(sh, 0.f, (size_t)NNODE * 2);
    k_fill_f32<<<(int)(NH / 256), 256, 0, stream>>>(conv, 0.f, NH);
    k_edge_max<<<NENT / 256, 256, 0, stream>>>(srcI, dstI, aes, aed, mh);
    k_edge_ex<<<NENT / 256, 256, 0, stream>>>(srcI, dstI, aes, aed, mh, exb, sh);
    k_conv_acc<<<NENT / 8, 256, 0, stream>>>(srcI, dstI, exb, sh, xw, conv);
    k_add_bias<<<(int)(NH / 256), 256, 0, stream>>>(conv, gatB + (size_t)l * HDIM, (int)NH);
    k_gemm_kn<2><<<2560, 256, 0, stream>>>(in, conv, gateWt + (size_t)l * 256 * 256,
                                           nullptr, xw, NNODE, HDIM, 2);       // gate -> xw
    k_gemm_kn<1><<<2560, 256, 0, stream>>>(conv, nullptr, fuseWt + (size_t)l * 256 * 128,
                                           nullptr, out, NNODE, HDIM, 1);      // tanh(conv@fw) -> out
    k_combine<<<(int)(NH / 256), 256, 0, stream>>>(xw, in, out, (int)NH);
  }

  // ---- masked global attention pooling ----
  k_gemm_kn<1><<<1280, 256, 0, stream>>>(bufA, nullptr, ph1Wt, ph1B, g1, NNODE, H2DIM, 1);
  k_pool_g<<<NNODE / 256, 256, 0, stream>>>(g1, pgW, pgB, hist, gsc);
  k_pooled<<<BDIM, HDIM, 0, stream>>>(gsc, bufA, pooled);

  // ---- fusion + vocab scores ----
  k_uw<<<2, 256, 0, stream>>>(dyn, pooled, wlW, wlB, uwv);
  k_pack_a<<<(int)(BDIM * 256 / 256), 256, 0, stream>>>(dyn, pooled, uwv, Abf);
  k_vocab<<<VDIM / 16, 256, 0, stream>>>(Abf, emb, gemb, fout);
}
